// TimeMixing_32727650795841
// MI455X (gfx1250) — compile-verified, matrix operand order checked by reading, not run on hardware
//
#include <hip/hip_runtime.h>

#define T_DIM 2048
#define C_DIM 2048
#define H_DIM 32
#define S_DIM 64
#define LORA_DIM 64
#define GLORA_DIM 128
#define EPS_GN 0.00064f

typedef __attribute__((ext_vector_type(16))) __bf16 v16bf;
typedef __attribute__((ext_vector_type(8)))  float  v8f;

union FragBF { uint4 u[2]; v16bf v; };

// Async global -> LDS 16B copy (CDNA5, ASYNCcnt-tracked, bypasses VGPRs).
__device__ __forceinline__ void async_copy16(void* lds_ptr, const void* gptr) {
  uint32_t l = (uint32_t)(uintptr_t)lds_ptr;   // LDS aperture: addr[31:0]
  uint64_t g = (uint64_t)(uintptr_t)gptr;
  asm volatile("global_load_async_to_lds_b128 %0, %1, off"
               :: "v"(l), "v"(g) : "memory");
}

// ---------------------------------------------------------------------------
// GEMM: C[MxN](fp32) = A[MxK](bf16, row-major) * Bt[NxK](bf16, pre-transposed)
// Block = 256 threads = 8 waves. Macro-tile 128(M) x 64(N), K step 32.
// Double-buffered async-LDS staging; each wave owns a 32x32 sub-block
// -> 4 v_wmma_f32_16x16x32_bf16 per K step.
// ---------------------------------------------------------------------------
__global__ __launch_bounds__(256) void gemm_bf16_wmma(
    const __bf16* __restrict__ A, const __bf16* __restrict__ Bt,
    float* __restrict__ C, int M, int N, int K)
{
  __shared__ __align__(16) __bf16 As[2][128][32];  // 2 x 8 KB
  __shared__ __align__(16) __bf16 Bs[2][64][32];   // 2 x 4 KB, [n][k]
  const int tid  = threadIdx.x;
  const int wv   = tid >> 5;
  const int lane = tid & 31;
  const int m0 = blockIdx.y * 128;
  const int n0 = blockIdx.x * 64;
  const int wm = (wv & 3) * 32;   // wave M offset in tile
  const int wn = (wv >> 2) * 32;  // wave N offset in tile
  const int srow = tid >> 2;      // staging row 0..63
  const int sseg = tid & 3;       // staging 16B segment
  const int khalf = lane >> 4;
  const int lr = lane & 15;

  v8f acc[2][2];
#pragma unroll
  for (int mi = 0; mi < 2; ++mi)
#pragma unroll
    for (int ni = 0; ni < 2; ++ni)
      acc[mi][ni] = (v8f){0.f, 0.f, 0.f, 0.f, 0.f, 0.f, 0.f, 0.f};

  // Issue the 3 async 16B copies for one K tile into buffer `buf`.
  auto stage = [&](int buf, int k0) {
    async_copy16(&As[buf][srow][sseg * 8],
                 A + (size_t)(m0 + srow) * K + k0 + sseg * 8);
    async_copy16(&As[buf][srow + 64][sseg * 8],
                 A + (size_t)(m0 + srow + 64) * K + k0 + sseg * 8);
    async_copy16(&Bs[buf][srow][sseg * 8],
                 Bt + (size_t)(n0 + srow) * K + k0 + sseg * 8);
  };

  const int nk = K >> 5;
  stage(0, 0);
  for (int kt = 0; kt < nk; ++kt) {
    const int cur = kt & 1;
    if (kt + 1 < nk) {
      stage(cur ^ 1, (kt + 1) << 5);
      // in-order completion: <=3 outstanding => current tile's copies landed
      asm volatile("s_wait_asynccnt 0x3" ::: "memory");
    } else {
      asm volatile("s_wait_asynccnt 0x0" ::: "memory");
    }
    __syncthreads();

    // CDNA5 16-bit 16x32 fragment layout: lane<16 row=lane K0..7/16..23,
    // lane>=16 K8..15/24..31 -> two 16B LDS reads per fragment.
    FragBF fa[2], fb[2];
#pragma unroll
    for (int mi = 0; mi < 2; ++mi) {
      const uint4* ra = (const uint4*)&As[cur][wm + mi * 16 + lr][0];
      fa[mi].u[0] = ra[khalf];
      fa[mi].u[1] = ra[khalf + 2];
    }
#pragma unroll
    for (int ni = 0; ni < 2; ++ni) {
      const uint4* rb = (const uint4*)&Bs[cur][wn + ni * 16 + lr][0];
      fb[ni].u[0] = rb[khalf];
      fb[ni].u[1] = rb[khalf + 2];
    }
#pragma unroll
    for (int mi = 0; mi < 2; ++mi)
#pragma unroll
      for (int ni = 0; ni < 2; ++ni)
        acc[mi][ni] = __builtin_amdgcn_wmma_f32_16x16x32_bf16(
            false, fa[mi].v, false, fb[ni].v, (short)0, acc[mi][ni],
            false, false);
    __syncthreads();  // all waves done reading buf[cur] before restage
  }

  // C/D layout: VGPR i -> M = i + 8*(lane>=16), N = lane%16
#pragma unroll
  for (int mi = 0; mi < 2; ++mi) {
#pragma unroll
    for (int ni = 0; ni < 2; ++ni) {
      const int row = m0 + wm + mi * 16 + khalf * 8;
      const int col = n0 + wn + ni * 16 + lr;
#pragma unroll
      for (int i = 0; i < 8; ++i)
        C[(size_t)(row + i) * N + col] = acc[mi][ni][i];
    }
  }
}

// ---------------------------------------------------------------------------
// Transpose + convert: out[n*K + k0..k0+7] = (bf16) in[(k0..k0+7)*N + n]
// grid((N+63)/64, K/8), block 64
// ---------------------------------------------------------------------------
__global__ void transpose_cvt_kernel(const float* __restrict__ in,
                                     __bf16* __restrict__ out, int K, int N)
{
  int n  = blockIdx.x * 64 + threadIdx.x;
  int k0 = blockIdx.y * 8;
  if (n >= N) return;
  union { __bf16 h[8]; uint4 u; } o;
#pragma unroll
  for (int i = 0; i < 8; ++i)
    o.h[i] = (__bf16)in[(size_t)(k0 + i) * N + n];
  *(uint4*)(out + (size_t)n * K + k0) = o.u;
}

// ---------------------------------------------------------------------------
// Token shift -> six bf16 mixed activations (GEMM A operands), 8-wide.
// ---------------------------------------------------------------------------
__global__ void token_shift_kernel(
    const float* __restrict__ x, const float* __restrict__ state,
    const unsigned char* __restrict__ new_starts,
    const float* cr, const float* cw, const float* ck,
    const float* cv, const float* ca, const float* cg,
    __bf16* xr, __bf16* xw, __bf16* xk, __bf16* xv, __bf16* xa, __bf16* xg)
{
  int idx = (blockIdx.x * blockDim.x + threadIdx.x) * 8;
  if (idx >= T_DIM * C_DIM) return;
  int t = idx >> 11;
  int c = idx & (C_DIM - 1);
  bool ns = new_starts[t] != 0;
  union O { __bf16 h[8]; uint4 u; };
  O or_, ow, ok, ov, oa, og;
#pragma unroll
  for (int i = 0; i < 8; ++i) {
    float xx = x[idx + i];
    float prev = (t == 0) ? state[c + i] : x[idx + i - C_DIM];
    if (ns) prev = 0.f;
    float sx = prev - xx;
    or_.h[i] = (__bf16)(xx + sx * cr[c + i]);
    ow.h[i]  = (__bf16)(xx + sx * cw[c + i]);
    ok.h[i]  = (__bf16)(xx + sx * ck[c + i]);
    ov.h[i]  = (__bf16)(xx + sx * cv[c + i]);
    oa.h[i]  = (__bf16)(xx + sx * ca[c + i]);
    og.h[i]  = (__bf16)(xx + sx * cg[c + i]);
  }
  *(uint4*)(xr + idx) = or_.u;
  *(uint4*)(xw + idx) = ow.u;
  *(uint4*)(xk + idx) = ok.u;
  *(uint4*)(xv + idx) = ov.u;
  *(uint4*)(xa + idx) = oa.u;
  *(uint4*)(xg + idx) = og.u;
}

__device__ __forceinline__ float sigmoidf_(float x) {
  return 1.f / (1.f + __expf(-x));
}

// fp32 -> bf16, 8-wide
__global__ void cvt_bf16_kernel(const float* __restrict__ in,
                                __bf16* __restrict__ out, int n) {
  int i = (blockIdx.x * blockDim.x + threadIdx.x) * 8;
  if (i >= n) return;
  union { __bf16 h[8]; uint4 u; } o;
#pragma unroll
  for (int j = 0; j < 8; ++j) o.h[j] = (__bf16)in[i + j];
  *(uint4*)(out + i) = o.u;
}

__global__ void tanh_cvt_kernel(const float* __restrict__ in,
                                __bf16* __restrict__ out, int n) {
  int i = (blockIdx.x * blockDim.x + threadIdx.x) * 8;
  if (i >= n) return;
  union { __bf16 h[8]; uint4 u; } o;
#pragma unroll
  for (int j = 0; j < 8; ++j) o.h[j] = (__bf16)tanhf(in[i + j]);
  *(uint4*)(out + i) = o.u;
}

__global__ void sigmoid_cvt_kernel(const float* __restrict__ in,
                                   __bf16* __restrict__ out, int n) {
  int i = (blockIdx.x * blockDim.x + threadIdx.x) * 8;
  if (i >= n) return;
  union { __bf16 h[8]; uint4 u; } o;
#pragma unroll
  for (int j = 0; j < 8; ++j) o.h[j] = (__bf16)sigmoidf_(in[i + j]);
  *(uint4*)(out + i) = o.u;
}

// w = -softplus(-(w0 + raw)) - 0.5, numerically stable (in place, fp32)
__global__ void w_transform_kernel(float* w, const float* w0, int n) {
  int i = blockIdx.x * blockDim.x + threadIdx.x;
  if (i >= n) return;
  float y = w0[i & (C_DIM - 1)] + w[i];
  float z = -y;
  float sp = fmaxf(z, 0.f) + log1pf(__expf(-fabsf(z)));
  w[i] = -sp - 0.5f;
}

// v = v + (v_first - v) * sigmoid(v0 + vlora)  (in place, fp32)
__global__ void v_mix_kernel(float* v, const float* vf, const float* v0,
                             const float* vlora, int n) {
  int i = blockIdx.x * blockDim.x + threadIdx.x;
  if (i >= n) return;
  float s = sigmoidf_(v0[i & (C_DIM - 1)] + vlora[i]);
  float vv = v[i];
  v[i] = vv + (vf[i] - vv) * s;
}

// a = sigmoid(a0 + raw)  (in place, fp32)
__global__ void a_transform_kernel(float* a, const float* a0, int n) {
  int i = blockIdx.x * blockDim.x + threadIdx.x;
  if (i >= n) return;
  a[i] = sigmoidf_(a0[i & (C_DIM - 1)] + a[i]);
}

// ---------------------------------------------------------------------------
// kk = normalize(k * k_k) per (t,h);  k <- k * (1 + (a-1)*k_a)   (in place)
// One wave32 per (t,h); lane handles s and s+32.
// ---------------------------------------------------------------------------
__global__ __launch_bounds__(32) void kk_prep_kernel(
    float* __restrict__ k, const float* __restrict__ a,
    const float* __restrict__ k_k, const float* __restrict__ k_a,
    float* __restrict__ kk)
{
  int g = blockIdx.x;                 // t*H + h
  int l = threadIdx.x;
  int base = g * S_DIM;
  int c0 = (g & (H_DIM - 1)) * S_DIM + l;
  int c1 = c0 + 32;
  float k0 = k[base + l], k1 = k[base + l + 32];
  float q0 = k0 * k_k[c0], q1 = k1 * k_k[c1];
  float ss = q0 * q0 + q1 * q1;
#pragma unroll
  for (int m = 16; m > 0; m >>= 1) ss += __shfl_xor(ss, m, 32);
  float inv = 1.f / fmaxf(sqrtf(ss), 1e-12f);
  kk[base + l]      = q0 * inv;
  kk[base + l + 32] = q1 * inv;
  float a0v = a[base + l], a1v = a[base + l + 32];
  k[base + l]      = k0 * (1.f + (a0v - 1.f) * k_a[c0]);
  k[base + l + 32] = k1 * (1.f + (a1v - 1.f) * k_a[c1]);
}

// ---------------------------------------------------------------------------
// Sequential RWKV7 state scan. One block per head; 64x64 state in registers
// (16 floats/thread), LDS for per-step vectors and cross-strip reductions.
// ---------------------------------------------------------------------------
__global__ __launch_bounds__(256) void scan_kernel(
    const float* __restrict__ r, const float* __restrict__ w,
    const float* __restrict__ k, const float* __restrict__ v,
    const float* __restrict__ kkb, const float* __restrict__ ab,
    const float* __restrict__ state, const unsigned char* __restrict__ new_starts,
    const int* __restrict__ length_p, float* __restrict__ y)
{
  const int h = blockIdx.x;
  const int tid = threadIdx.x;
  const int i = tid & 63;       // state row
  const int strip = tid >> 6;   // 0..3
  const int j0 = strip * 16;    // state col base

  float s[16];
  const float* sbase = state + C_DIM + (size_t)h * S_DIM * S_DIM;
#pragma unroll
  for (int e = 0; e < 16; ++e) s[e] = sbase[i * S_DIM + j0 + e];

  __shared__ float r_s[64], we_s[64], k_s[64], v_s[64], aa_s[64], bb_s[64];
  __shared__ float part[4][64];
  __shared__ float sab_s[64];
  __shared__ float part2[4][64];
  const int length = length_p[0];

  for (int t = 0; t < T_DIM; ++t) {
    const int base = t * C_DIM + h * S_DIM;
    if (tid < 64) {
      r_s[tid]  = r[base + tid];
      we_s[tid] = __expf(w[base + tid]);
      k_s[tid]  = k[base + tid];
      v_s[tid]  = v[base + tid];
      float kkv = kkb[base + tid];
      float av  = ab[base + tid];
      aa_s[tid] = -kkv;
      bb_s[tid] = kkv * av;
    }
    const bool ns = new_starts[t] != 0;
    __syncthreads();

    // sab = S_in @ a   (partial dot over this strip's 16 columns)
    float pa = 0.f;
#pragma unroll
    for (int e = 0; e < 16; ++e) {
      float se = ns ? 0.f : s[e];
      pa += se * aa_s[j0 + e];
    }
    part[strip][i] = pa;
    __syncthreads();
    if (tid < 64)
      sab_s[tid] = part[0][tid] + part[1][tid] + part[2][tid] + part[3][tid];
    __syncthreads();

    const bool act = t < length;
    const float vi = v_s[i];
    const float sabi = sab_s[i];
    float po = 0.f;
#pragma unroll
    for (int e = 0; e < 16; ++e) {
      int j = j0 + e;
      float se = ns ? 0.f : s[e];
      float snew = se * we_s[j] + sabi * bb_s[j] + vi * k_s[j];
      po += snew * r_s[j];
      if (act) s[e] = snew;   // freeze state past `length`
    }
    part2[strip][i] = po;
    __syncthreads();
    if (tid < 64)
      y[base + tid] = part2[0][tid] + part2[1][tid] + part2[2][tid] + part2[3][tid];
    __syncthreads();
  }
}

// ---------------------------------------------------------------------------
// GroupNorm (groups=H, eps) + bonus rkv term + gate; bf16 output (feeds GEMM).
// Wave per (t,h).
// ---------------------------------------------------------------------------
__global__ __launch_bounds__(32) void gn_bonus_gate_kernel(
    const float* __restrict__ y, const float* __restrict__ r,
    const float* __restrict__ k, const float* __restrict__ v,
    const float* __restrict__ g, const float* __restrict__ r_k,
    const float* __restrict__ ln_w, const float* __restrict__ ln_b,
    __bf16* __restrict__ y2)
{
  int grp = blockIdx.x;            // t*H + h
  int l = threadIdx.x;
  int base = grp * S_DIM;
  int c0 = (grp & (H_DIM - 1)) * S_DIM + l;   // h*64 + s
  int c1 = c0 + 32;
  float y0 = y[base + l], y1 = y[base + l + 32];
  float s1 = y0 + y1;
  float s2 = y0 * y0 + y1 * y1;
  float dot = r[base + l] * k[base + l] * r_k[c0]
            + r[base + l + 32] * k[base + l + 32] * r_k[c1];
#pragma unroll
  for (int m = 16; m > 0; m >>= 1) {
    s1  += __shfl_xor(s1, m, 32);
    s2  += __shfl_xor(s2, m, 32);
    dot += __shfl_xor(dot, m, 32);
  }
  float mu  = s1 * (1.f / S_DIM);
  float var = s2 * (1.f / S_DIM) - mu * mu;
  float inv = rsqrtf(var + EPS_GN);
  float n0 = (y0 - mu) * inv * ln_w[c0] + ln_b[c0];
  float n1 = (y1 - mu) * inv * ln_w[c1] + ln_b[c1];
  y2[base + l]      = (__bf16)((n0 + dot * v[base + l])      * g[base + l]);
  y2[base + l + 32] = (__bf16)((n1 + dot * v[base + l + 32]) * g[base + l + 32]);
}

// ---------------------------------------------------------------------------
extern "C" void kernel_launch(void* const* d_in, const int* in_sizes, int n_in,
                              void* d_out, int out_size, void* d_ws, size_t ws_size,
                              hipStream_t stream) {
  (void)in_sizes; (void)n_in; (void)out_size; (void)ws_size;
  const float* x        = (const float*)d_in[0];
  const float* state    = (const float*)d_in[1];
  const float* v_first  = (const float*)d_in[2];
  const unsigned char* new_starts = (const unsigned char*)d_in[3];
  const int*   length   = (const int*)d_in[4];
  const float* x_r = (const float*)d_in[5];
  const float* x_w = (const float*)d_in[6];
  const float* x_k = (const float*)d_in[7];
  const float* x_v = (const float*)d_in[8];
  const float* x_a = (const float*)d_in[9];
  const float* x_g = (const float*)d_in[10];
  const float* W_r = (const float*)d_in[11];
  const float* W_k = (const float*)d_in[12];
  const float* W_v = (const float*)d_in[13];
  const float* W_o = (const float*)d_in[14];
  const float* w0  = (const float*)d_in[15];
  const float* w1  = (const float*)d_in[16];
  const float* w2  = (const float*)d_in[17];
  const float* v0  = (const float*)d_in[18];
  const float* v1  = (const float*)d_in[19];
  const float* v2  = (const float*)d_in[20];
  const float* a0  = (const float*)d_in[21];
  const float* a1  = (const float*)d_in[22];
  const float* a2  = (const float*)d_in[23];
  const float* g1  = (const float*)d_in[24];
  const float* g2  = (const float*)d_in[25];
  const float* k_k = (const float*)d_in[26];
  const float* k_a = (const float*)d_in[27];
  const float* r_k = (const float*)d_in[28];
  const float* ln_w = (const float*)d_in[29];
  const float* ln_b = (const float*)d_in[30];
  float* out = (float*)d_out;

  const size_t TC = (size_t)T_DIM * C_DIM;
  char* wsb = (char*)d_ws;
  auto allocF = [&](size_t n) { float* p = (float*)wsb; wsb += n * sizeof(float); return p; };
  auto allocH = [&](size_t n) { __bf16* p = (__bf16*)wsb; wsb += n * sizeof(__bf16); return p; };

  // fp32 intermediates
  float* rb   = allocF(TC);
  float* wb   = allocF(TC);
  float* kb   = allocF(TC);
  float* vb   = allocF(TC);
  float* ab   = allocF(TC);
  float* gb   = allocF(TC);
  float* kkb  = allocF(TC);
  float* vtmp = allocF(TC);
  float* t64f  = allocF((size_t)T_DIM * LORA_DIM);
  float* t128f = allocF((size_t)T_DIM * GLORA_DIM);
  // bf16 GEMM operands
  __bf16* xr = allocH(TC); __bf16* xw = allocH(TC); __bf16* xk = allocH(TC);
  __bf16* xv = allocH(TC); __bf16* xa = allocH(TC); __bf16* xg = allocH(TC);
  __bf16* Wrt = allocH(TC); __bf16* Wkt = allocH(TC);
  __bf16* Wvt = allocH(TC); __bf16* Wot = allocH(TC);
  __bf16* w1t = allocH((size_t)LORA_DIM * C_DIM);
  __bf16* w2t = allocH((size_t)C_DIM * LORA_DIM);
  __bf16* v1t = allocH((size_t)LORA_DIM * C_DIM);
  __bf16* v2t = allocH((size_t)C_DIM * LORA_DIM);
  __bf16* a1t = allocH((size_t)LORA_DIM * C_DIM);
  __bf16* a2t = allocH((size_t)C_DIM * LORA_DIM);
  __bf16* g1t = allocH((size_t)GLORA_DIM * C_DIM);
  __bf16* g2t = allocH((size_t)C_DIM * GLORA_DIM);
  __bf16* t64b  = allocH((size_t)T_DIM * LORA_DIM);
  __bf16* t128b = allocH((size_t)T_DIM * GLORA_DIM);
  __bf16* y2b   = allocH(TC);
  float* yb = vtmp;  // scan output reuses vtmp (free after v_mix)

  const int total = T_DIM * C_DIM;
  const int EB = 256;
  const int EG  = (total + EB - 1) / EB;
  const int EG8 = (total / 8 + EB - 1) / EB;

  auto gemm = [&](const __bf16* A, const __bf16* Bt, float* C, int M, int N, int K) {
    dim3 grid(N / 64, M / 128);
    gemm_bf16_wmma<<<grid, 256, 0, stream>>>(A, Bt, C, M, N, K);
  };
  auto tcvt = [&](const float* B, __bf16* Bt, int K, int N) {
    dim3 grid((N + 63) / 64, K / 8);
    transpose_cvt_kernel<<<grid, 64, 0, stream>>>(B, Bt, K, N);
  };

  // 0. weights -> transposed bf16 (N x K)
  tcvt(W_r, Wrt, C_DIM, C_DIM);
  tcvt(W_k, Wkt, C_DIM, C_DIM);
  tcvt(W_v, Wvt, C_DIM, C_DIM);
  tcvt(W_o, Wot, C_DIM, C_DIM);
  tcvt(w1, w1t, C_DIM, LORA_DIM);
  tcvt(w2, w2t, LORA_DIM, C_DIM);
  tcvt(v1, v1t, C_DIM, LORA_DIM);
  tcvt(v2, v2t, LORA_DIM, C_DIM);
  tcvt(a1, a1t, C_DIM, LORA_DIM);
  tcvt(a2, a2t, LORA_DIM, C_DIM);
  tcvt(g1, g1t, C_DIM, GLORA_DIM);
  tcvt(g2, g2t, GLORA_DIM, C_DIM);

  // 1. token shift -> bf16 mixed activations
  token_shift_kernel<<<EG8, EB, 0, stream>>>(x, state, new_starts,
      x_r, x_w, x_k, x_v, x_a, x_g, xr, xw, xk, xv, xa, xg);

  // 2. big projections
  gemm(xr, Wrt, rb, T_DIM, C_DIM, C_DIM);
  gemm(xk, Wkt, kb, T_DIM, C_DIM, C_DIM);
  gemm(xv, Wvt, vb, T_DIM, C_DIM, C_DIM);

  // 3. decay LoRA: w = -softplus(-(w0 + tanh(xw@w1)@w2)) - 0.5
  gemm(xw, w1t, t64f, T_DIM, LORA_DIM, C_DIM);
  tanh_cvt_kernel<<<(T_DIM * LORA_DIM / 8 + EB - 1) / EB, EB, 0, stream>>>(
      t64f, t64b, T_DIM * LORA_DIM);
  gemm(t64b, w2t, wb, T_DIM, C_DIM, LORA_DIM);
  w_transform_kernel<<<EG, EB, 0, stream>>>(wb, w0, total);

  // 4. value-residual LoRA
  gemm(xv, v1t, t64f, T_DIM, LORA_DIM, C_DIM);
  cvt_bf16_kernel<<<(T_DIM * LORA_DIM / 8 + EB - 1) / EB, EB, 0, stream>>>(
      t64f, t64b, T_DIM * LORA_DIM);
  gemm(t64b, v2t, vtmp, T_DIM, C_DIM, LORA_DIM);
  v_mix_kernel<<<EG, EB, 0, stream>>>(vb, v_first, v0, vtmp, total);

  // 5. in-context-lr LoRA
  gemm(xa, a1t, t64f, T_DIM, LORA_DIM, C_DIM);
  cvt_bf16_kernel<<<(T_DIM * LORA_DIM / 8 + EB - 1) / EB, EB, 0, stream>>>(
      t64f, t64b, T_DIM * LORA_DIM);
  gemm(t64b, a2t, ab, T_DIM, C_DIM, LORA_DIM);
  a_transform_kernel<<<EG, EB, 0, stream>>>(ab, a0, total);

  // 6. gate
  gemm(xg, g1t, t128f, T_DIM, GLORA_DIM, C_DIM);
  sigmoid_cvt_kernel<<<(T_DIM * GLORA_DIM / 8 + EB - 1) / EB, EB, 0, stream>>>(
      t128f, t128b, T_DIM * GLORA_DIM);
  gemm(t128b, g2t, gb, T_DIM, C_DIM, GLORA_DIM);

  // 7. kk normalize + k_a blend (in place on kb)
  kk_prep_kernel<<<T_DIM * H_DIM, 32, 0, stream>>>(kb, ab, k_k, k_a, kkb);

  // 8. sequential scan
  scan_kernel<<<H_DIM, 256, 0, stream>>>(rb, wb, kb, vb, kkb, ab,
                                         state, new_starts, length, yb);

  // 9. groupnorm + bonus + gate -> bf16
  gn_bonus_gate_kernel<<<T_DIM * H_DIM, 32, 0, stream>>>(yb, rb, kb, vb, gb,
                                                         r_k, ln_w, ln_b, y2b);

  // 10. output projection
  gemm(y2b, Wot, out, T_DIM, C_DIM, C_DIM);
}
